// FluidLayer_6768868459069
// MI455X (gfx1250) — matrix operand order repaired and checked by reading, not compiled, not run on hardware
//
#include <hip/hip_runtime.h>

typedef __attribute__((ext_vector_type(16))) __bf16 v16bf;
typedef __attribute__((ext_vector_type(8)))  float  v8f;

#define HWSZ 65536   // 256*256
#define WID  256

// ---------------------------------------------------------------------------
// 1) NCHW f32 -> NHWC bf16 (LDS tile transpose, coalesced reads & writes)
// ---------------------------------------------------------------------------
__global__ __launch_bounds__(256) void k_xcvt(const float* __restrict__ x,
                                              __bf16* __restrict__ xb) {
  __shared__ __bf16 tile[64][65];               // [pixel][channel], padded
  const int b  = blockIdx.y;
  const int p0 = blockIdx.x * 64;               // 64 consecutive pixels
  const int tp = threadIdx.x & 63;
  const int t4 = threadIdx.x >> 6;              // 0..3
  const float* xin = x + (size_t)b * 64 * HWSZ + p0;
  for (int c = t4; c < 64; c += 4)
    tile[tp][c] = (__bf16)xin[(size_t)c * HWSZ + tp];
  __syncthreads();
  __bf16* xo = xb + ((size_t)b * HWSZ + p0) * 64;
  for (int pp = t4; pp < 64; pp += 4)
    xo[(size_t)pp * 64 + tp] = tile[pp][tp];
}

// ---------------------------------------------------------------------------
// 2) Pack w_int (OIHW f32) into WMMA A-fragment layout: wA[4][18][32][16] bf16
//    K-step t covers ONE tap: tap = t>>1 ; ci = (t&1)*32 + k32
//    Contiguous k order within step, same as B: k32 = (half<<4) + e
// ---------------------------------------------------------------------------
__global__ __launch_bounds__(256) void k_packw(const float* __restrict__ w_int,
                                               __bf16* __restrict__ wA) {
  for (int idx = threadIdx.x; idx < 4 * 18 * 32 * 16; idx += 256) {
    int e     = idx & 15;
    int lane  = (idx >> 4) & 31;
    int t     = (idx >> 9) % 18;
    int mtile = idx / (18 * 512);
    int M     = mtile * 16 + (lane & 15);
    int half  = lane >> 4;
    int k32   = (half << 4) + e;
    int tap   = t >> 1;
    int ci    = ((t & 1) << 5) + k32;
    int kh    = tap / 3, kw = tap - kh * 3;
    wA[idx] = (__bf16)w_int[((M * 64 + ci) * 3 + kh) * 3 + kw];
  }
}

// ---------------------------------------------------------------------------
// 3) Implicit-GEMM conv with WMMA bf16. Block = 8 waves; each wave does
//    one 16-channel M-tile x two 16-pixel N-tiles (A fragment reused).
//    Every pixel uses w_int (+bias); borders fixed later.
// ---------------------------------------------------------------------------
__global__ __launch_bounds__(256) void k_conv_wmma(const __bf16* __restrict__ xb,
                                                   const __bf16* __restrict__ wA,
                                                   const float*  __restrict__ bias,
                                                   float* __restrict__ y) {
  const int lane  = threadIdx.x & 31;
  const int wave  = threadIdx.x >> 5;
  const int mtile = wave & 3;
  const int npair = wave >> 2;                    // 0..1
  const int b     = blockIdx.y;
  const int p0    = blockIdx.x * 64 + npair * 32; // stays within one row
  const int i     = p0 >> 8;
  const int ln    = lane & 15;
  const int half  = lane >> 4;
  const int j0    = (p0 & 255) + ln;
  const int j1    = j0 + 16;                      // j1 >= 16, never 0
  const int r0  = (i == 0) ? 253 : ((i == 255) ? 0 : i - 1);
  const int c0a = (j0 == 0) ? 0 : ((j0 == 255) ? 253 : j0 - 1);
  const int c0b = (j1 == 255) ? 253 : j1 - 1;

  const __bf16* xbase = xb + (size_t)b * (HWSZ * 64);
  int off[9];
#pragma unroll
  for (int tap = 0; tap < 9; ++tap) {
    int kh = tap / 3, kw = tap - kh * 3;
    off[tap] = ((((r0 + kh) << 8) + (c0a + kw)) << 6) + (half << 4);
  }
  const int delta = (c0b - c0a) << 6;
  const __bf16* wAl = wA + ((mtile * 18 * 32 + lane) << 4);

  v8f acc0 = {}, acc1 = {};
#pragma unroll
  for (int t = 0; t < 18; ++t) {
    const int tap = t >> 1;
    const int cio = (t & 1) << 5;                 // 0 or 32 elements
    v16bf a  = *(const v16bf*)(wAl + (t << 9));
    v16bf b0 = *(const v16bf*)(xbase + off[tap] + cio);
    v16bf b1 = *(const v16bf*)(xbase + off[tap] + delta + cio);
    acc0 = __builtin_amdgcn_wmma_f32_16x16x32_bf16(false, a, false, b0,
                                                   (short)0, acc0, false, false);
    acc1 = __builtin_amdgcn_wmma_f32_16x16x32_bf16(false, a, false, b1,
                                                   (short)0, acc1, false, false);
  }

  float* ybase = y + (size_t)b * (64 * HWSZ);
  const int pA = p0 + ln, pB = pA + 16;
#pragma unroll
  for (int r = 0; r < 8; ++r) {
    int co = mtile * 16 + half * 8 + r;           // C/D layout: M = r + 8*half
    float bs = bias[co];
    ybase[(size_t)co * HWSZ + pA] = acc0[r] + bs;
    ybase[(size_t)co * HWSZ + pB] = acc1[r] + bs;
  }
}

// ---------------------------------------------------------------------------
// 4) Border fixup: recompute 1020 border pixels/batch with correct weights
// ---------------------------------------------------------------------------
__global__ __launch_bounds__(256) void k_border(
    const float* __restrict__ x,
    const float* __restrict__ w_tl, const float* __restrict__ w_tr,
    const float* __restrict__ w_bl, const float* __restrict__ w_br,
    const float* __restrict__ w_top, const float* __restrict__ w_bot,
    const float* __restrict__ w_left, const float* __restrict__ w_right,
    const float* __restrict__ bias, float* __restrict__ y) {
  int u = blockIdx.x * 256 + threadIdx.x;
  if (u >= 8 * 64 * 1020) return;
  int px = u % 1020;
  int co = (u / 1020) & 63;
  int b  = u / (1020 * 64);
  int i, j;
  if (px < 256)      { i = 0;   j = px; }
  else if (px < 512) { i = 255; j = px - 256; }
  else if (px < 766) { j = 0;   i = px - 511; }
  else               { j = 255; i = px - 765; }
  const float* w;
  if (i == 0)        w = (j == 0) ? w_bl : (j == 255) ? w_br : w_bot;
  else if (i == 255) w = (j == 0) ? w_tl : (j == 255) ? w_tr : w_top;
  else               w = (j == 0) ? w_left : w_right;
  int r0 = (i == 0) ? 253 : (i == 255) ? 0 : i - 1;
  int c0 = (j == 0) ? 0 : (j == 255) ? 253 : j - 1;
  const float* xb_  = x + (size_t)b * 64 * HWSZ;
  const float* wrow = w + co * 576;
  float s = 0.f;
  for (int ci = 0; ci < 64; ++ci) {
    const float* xp = xb_ + (size_t)ci * HWSZ + r0 * WID + c0;
    const float* wp = wrow + ci * 9;
#pragma unroll
    for (int kh = 0; kh < 3; ++kh)
#pragma unroll
      for (int kw = 0; kw < 3; ++kw)
        s += xp[kh * WID + kw] * wp[kh * 3 + kw];
  }
  y[(size_t)(b * 64 + co) * HWSZ + i * WID + j] = s + bias[co];
}

// ---------------------------------------------------------------------------
// 5) GroupNorm stats: one block per (batch,group); segment is contiguous
// ---------------------------------------------------------------------------
__global__ __launch_bounds__(256) void k_stats(const float* __restrict__ y,
                                               float* __restrict__ stats) {
  const int bg = blockIdx.x;                       // 0..127
  const float* p = y + (size_t)bg * 262144;
  float s = 0.f, ss = 0.f;
  for (int k = threadIdx.x; k < 262144; k += 256) {
    float v = p[k]; s += v; ss += v * v;
  }
  __shared__ float sh[512];
  sh[threadIdx.x] = s; sh[256 + threadIdx.x] = ss;
  __syncthreads();
  for (int st = 128; st > 0; st >>= 1) {
    if (threadIdx.x < st) {
      sh[threadIdx.x]       += sh[threadIdx.x + st];
      sh[256 + threadIdx.x] += sh[256 + threadIdx.x + st];
    }
    __syncthreads();
  }
  if (threadIdx.x == 0) {
    float mu  = sh[0] * (1.f / 262144.f);
    float var = sh[256] * (1.f / 262144.f) - mu * mu;
    stats[2 * bg]     = mu;
    stats[2 * bg + 1] = rsqrtf(var + 1e-5f);
  }
}

// ---------------------------------------------------------------------------
// 6) Normalize + affine + SELU, in place on d_out
// ---------------------------------------------------------------------------
__global__ __launch_bounds__(256) void k_norm(float* __restrict__ y,
                                              const float* __restrict__ stats,
                                              const float* __restrict__ gamma,
                                              const float* __restrict__ beta) {
  unsigned idx = blockIdx.x * 256u + threadIdx.x;   // < 33554432
  int c  = (idx >> 16) & 63;
  int bg = idx >> 18;                               // / 262144
  float mu = stats[2 * bg], rs = stats[2 * bg + 1];
  float v  = (y[idx] - mu) * rs * gamma[c] + beta[c];
  const float SC = 1.0507009873554805f, AL = 1.6732632423543772f;
  y[idx] = v > 0.f ? SC * v : SC * AL * (__expf(v) - 1.f);
}

// ---------------------------------------------------------------------------
extern "C" void kernel_launch(void* const* d_in, const int* in_sizes, int n_in,
                              void* d_out, int out_size, void* d_ws, size_t ws_size,
                              hipStream_t stream) {
  const float* x       = (const float*)d_in[0];
  const float* w_int   = (const float*)d_in[1];
  const float* w_tl    = (const float*)d_in[2];
  const float* w_tr    = (const float*)d_in[3];
  const float* w_bl    = (const float*)d_in[4];
  const float* w_br    = (const float*)d_in[5];
  const float* w_top   = (const float*)d_in[6];
  const float* w_bot   = (const float*)d_in[7];
  const float* w_left  = (const float*)d_in[8];
  const float* w_right = (const float*)d_in[9];
  const float* bias    = (const float*)d_in[10];
  const float* gamma   = (const float*)d_in[11];
  const float* beta    = (const float*)d_in[12];
  float* y = (float*)d_out;

  // ws layout: [0, 64MiB) x NHWC bf16 | wA bf16 (73728B) | stats (1KiB)
  __bf16* xb    = (__bf16*)d_ws;
  __bf16* wA    = (__bf16*)((char*)d_ws + 67108864);
  float*  stats = (float*)((char*)d_ws + 67108864 + 73728);

  k_xcvt     <<<dim3(1024, 8), 256, 0, stream>>>(x, xb);
  k_packw    <<<1,            256, 0, stream>>>(w_int, wA);
  k_conv_wmma<<<dim3(1024, 8), 256, 0, stream>>>(xb, wA, bias, y);
  k_border   <<<2040,         256, 0, stream>>>(x, w_tl, w_tr, w_bl, w_br,
                                                w_top, w_bot, w_left, w_right,
                                                bias, y);
  k_stats    <<<128,          256, 0, stream>>>(y, stats);
  k_norm     <<<131072,       256, 0, stream>>>(y, stats, gamma, beta);
}
